// ODEBlock_41635412967613
// MI455X (gfx1250) — compile-verified, matrix operand order checked by reading, not run on hardware
//
#include <hip/hip_runtime.h>
#include <hip/hip_bf16.h>

// CNF ODE (FFJORD-style) fully fused for MI455X / gfx1250, wave32.
// Each workgroup integrates 16 independent samples through all RK4 steps.
// Dominant cost: per RHS eval, a (64 x 512) @ (512 x 512) GEMM done with
// v_wmma_f32_16x16x32_f16 (f16 in, f32 accumulate). W1 is packed once into
// WMMA-B fragment order so every lane load is a contiguous 32B global load.

typedef __attribute__((ext_vector_type(16))) _Float16 v16h;
typedef __attribute__((ext_vector_type(8)))  _Float16 v8h;
typedef __attribute__((ext_vector_type(8)))  float    v8f;

constexpr int HID    = 512;        // hidden width
constexpr int DIM    = 3;          // data dim
constexpr int NSTEP  = 48;         // fixed RK4 steps over [0,1]
constexpr int SPW    = 16;         // samples per workgroup
constexpr int XROWS  = 64;         // 16 samples x (h + 3 tangents)
constexpr int KTILES = HID / 32;   // 16 K-steps of 32
constexpr float T0C  = 1.0f;

// ---------------------------------------------------------------------------
// Pack W1 (f32 row-major HxH) into f16 WMMA-B fragments.
// Fragment order: W1p[(((nt*16)+kt)*32 + lane)*16 + j], nt: N-tile of 16,
// kt: K-tile of 32. Lane<16 holds column n=nt*16+lane with K = kt*32+{0..7}
// (j=0..7) and kt*32+{16..23} (j=8..15); lanes 16..31 hold the +8 K halves,
// mirroring the documented 16-bit A-matrix layout.
// ---------------------------------------------------------------------------
__global__ void pack_w1(const float* __restrict__ W1, _Float16* __restrict__ W1p) {
    int gid = blockIdx.x * blockDim.x + threadIdx.x;   // 0 .. HID*HID-1
    int j    = gid & 15;
    int L    = (gid >> 4) & 31;
    int kt   = (gid >> 9) & 15;
    int nt   = gid >> 13;
    int n    = nt * 16 + (L & 15);
    int half = L >> 4;
    int vg = j >> 1, lo = j & 1;
    int k_local = (vg < 4) ? (vg * 2 + lo + half * 8)
                           : (16 + (vg - 4) * 2 + lo + half * 8);
    int k = kt * 32 + k_local;
    W1p[gid] = (_Float16)W1[k * HID + n];
}

// ---------------------------------------------------------------------------
// One RHS evaluation: (dz, -tr J) for 16 samples.
//   ys[s][0..2] = x, t = scalar time.  Result -> kk[s][0..3].
// ---------------------------------------------------------------------------
__device__ __forceinline__ void rhs_eval(
    float t,
    _Float16* __restrict__ Xs,
    float (*__restrict__ ys)[4], float (*__restrict__ kk)[4],
    float (*__restrict__ red)[16][6],
    const float* __restrict__ W0, const float* __restrict__ b0,
    const float* __restrict__ b1,
    const float* __restrict__ Wl, const float* __restrict__ bl,
    const _Float16* __restrict__ W1p,
    int tid, int wave, int lane)
{
    // ---- layer 1 + constant input tangents [0,e_i] (tiny; VALU) ----------
    #pragma unroll 4
    for (int i = 0; i < 32; ++i) {
        int idx = tid * 32 + i;            // 0..8191 = 16 samples x 512
        int s   = idx >> 9;
        int n   = idx & (HID - 1);
        float w1r = W0[1 * HID + n];
        float w2r = W0[2 * HID + n];
        float w3r = W0[3 * HID + n];
        float a0 = t * W0[n] + ys[s][0] * w1r + ys[s][1] * w2r
                 + ys[s][2] * w3r + b0[n];
        float h  = tanhf(a0);
        float dt = 1.0f - h * h;           // tanh'
        Xs[( 0 + s) * HID + n] = (_Float16)h;
        Xs[(16 + s) * HID + n] = (_Float16)(dt * w1r);
        Xs[(32 + s) * HID + n] = (_Float16)(dt * w2r);
        Xs[(48 + s) * HID + n] = (_Float16)(dt * w3r);
    }
    __syncthreads();

    // ---- layer 2 GEMM: (64 x 512) @ (512 x 512), WMMA f16->f32 ----------
    int lanem   = lane & 15;
    int halfsel = lane >> 4;
    v8f acc[4][4];
    #pragma unroll
    for (int mt = 0; mt < 4; ++mt)
        #pragma unroll
        for (int ntl = 0; ntl < 4; ++ntl)
            acc[mt][ntl] = (v8f){0.f,0.f,0.f,0.f,0.f,0.f,0.f,0.f};

    for (int kt = 0; kt < KTILES; ++kt) {
        // A fragments from LDS (two contiguous 16B halves per lane)
        v16h a[4];
        int kbase = kt * 32 + halfsel * 8;
        #pragma unroll
        for (int mt = 0; mt < 4; ++mt) {
            const _Float16* p = Xs + (mt * 16 + lanem) * HID + kbase;
            v8h lo = *(const v8h*)(p);
            v8h hi = *(const v8h*)(p + 16);
            a[mt] = __builtin_shufflevector(lo, hi,
                        0,1,2,3,4,5,6,7,8,9,10,11,12,13,14,15);
        }
        // B fragments: one contiguous 32B global load per lane per N-tile
        v16h b[4];
        #pragma unroll
        for (int ntl = 0; ntl < 4; ++ntl) {
            int nt = wave * 4 + ntl;
            b[ntl] = *(const v16h*)(W1p + (((nt * 16) + kt) * 32 + lane) * 16);
        }
        if (kt + 1 < KTILES)   // pull next K-step of W1 toward the WGP
            __builtin_prefetch(W1p + (((wave * 4 * 16) + kt + 1) * 32 + lane) * 16, 0, 3);
        #pragma unroll
        for (int mt = 0; mt < 4; ++mt)
            #pragma unroll
            for (int ntl = 0; ntl < 4; ++ntl)
                acc[mt][ntl] = __builtin_amdgcn_wmma_f32_16x16x32_f16(
                    false, a[mt], false, b[ntl],
                    (short)0, acc[mt][ntl], false, false);
    }
    __syncthreads();   // everyone finished reading Xs

    // ---- epilogue: tanh + tangent rescale, write next activations --------
    #pragma unroll
    for (int ntl = 0; ntl < 4; ++ntl) {
        int n = (wave * 4 + ntl) * 16 + lanem;
        float bn = b1[n];
        #pragma unroll
        for (int r = 0; r < 8; ++r) {
            int m   = r + 8 * halfsel;     // sample 0..15
            float a1 = acc[0][ntl][r] + bn;
            float h  = tanhf(a1);
            float dt = 1.0f - h * h;
            Xs[( 0 + m) * HID + n] = (_Float16)h;
            Xs[(16 + m) * HID + n] = (_Float16)(dt * acc[1][ntl][r]);
            Xs[(32 + m) * HID + n] = (_Float16)(dt * acc[2][ntl][r]);
            Xs[(48 + m) * HID + n] = (_Float16)(dt * acc[3][ntl][r]);
        }
    }
    __syncthreads();

    // ---- layer 3 + Jacobian trace (tiny; VALU + LDS reduce) --------------
    {
        int s = tid >> 4;                  // sample
        int c = tid & 15;                  // feature chunk of 32
        float p0=0.f,p1=0.f,p2=0.f,p3=0.f,p4=0.f,p5=0.f;
        int f0 = c * 32;
        #pragma unroll 4
        for (int f = f0; f < f0 + 32; ++f) {
            float h  = (float)Xs[( 0 + s) * HID + f];
            float u1 = (float)Xs[(16 + s) * HID + f];
            float u2 = (float)Xs[(32 + s) * HID + f];
            float u3 = (float)Xs[(48 + s) * HID + f];
            float wa = Wl[f * DIM + 0];
            float wb = Wl[f * DIM + 1];
            float wc = Wl[f * DIM + 2];
            p0 += h * wa;  p1 += h * wb;  p2 += h * wc;
            p3 += u1 * wa; p4 += u2 * wb; p5 += u3 * wc;   // J_ii pieces
        }
        red[s][c][0]=p0; red[s][c][1]=p1; red[s][c][2]=p2;
        red[s][c][3]=p3; red[s][c][4]=p4; red[s][c][5]=p5;
    }
    __syncthreads();
    if (tid < SPW) {
        int s = tid;
        float q0=0.f,q1=0.f,q2=0.f,q3=0.f,q4=0.f,q5=0.f;
        #pragma unroll
        for (int c = 0; c < 16; ++c) {
            q0+=red[s][c][0]; q1+=red[s][c][1]; q2+=red[s][c][2];
            q3+=red[s][c][3]; q4+=red[s][c][4]; q5+=red[s][c][5];
        }
        kk[s][0] = T0C * (q0 + bl[0]);
        kk[s][1] = T0C * (q1 + bl[1]);
        kk[s][2] = T0C * (q2 + bl[2]);
        kk[s][3] = T0C * (-(q3 + q4 + q5));   // dlogp = -tr J
    }
    __syncthreads();
}

// ---------------------------------------------------------------------------
// Fused fixed-step RK4 over [0,1]; 16 samples per workgroup, 8 wave32 waves.
// ---------------------------------------------------------------------------
__global__ __launch_bounds__(256, 1)
void cnf_rk4(const float* __restrict__ states,
             const float* __restrict__ W0, const float* __restrict__ b0,
             const float* __restrict__ b1,
             const float* __restrict__ Wl, const float* __restrict__ bl,
             const _Float16* __restrict__ W1p,
             float* __restrict__ out)
{
    __shared__ _Float16 Xs[XROWS * HID];           // 64 KB activations
    __shared__ float ys [SPW][4];                  // stage state
    __shared__ float y0s[SPW][4];                  // step-start state
    __shared__ float kk [SPW][4];                  // RHS result
    __shared__ float av [SPW][4];                  // RK accumulator
    __shared__ float red[SPW][16][6];              // layer-3 reduction

    int tid  = threadIdx.x;
    int wave = tid >> 5;
    int lane = tid & 31;
    int sbase = blockIdx.x * SPW;

    if (tid < SPW * 4) {
        int s = tid >> 2, c = tid & 3;
        y0s[s][c] = states[(sbase + s) * 4 + c];
    }
    __syncthreads();

    const float hs = 1.0f / (float)NSTEP;
    for (int step = 0; step < NSTEP; ++step) {
        float t0 = (float)step * hs;
        // k1
        if (tid < SPW * 4) { int s=tid>>2, c=tid&3; ys[s][c] = y0s[s][c]; }
        __syncthreads();
        rhs_eval(t0, Xs, ys, kk, red, W0, b0, b1, Wl, bl, W1p, tid, wave, lane);
        if (tid < SPW * 4) { int s=tid>>2, c=tid&3;
            av[s][c] = y0s[s][c] + (hs/6.0f)*kk[s][c];
            ys[s][c] = y0s[s][c] + 0.5f*hs*kk[s][c]; }
        __syncthreads();
        // k2
        rhs_eval(t0 + 0.5f*hs, Xs, ys, kk, red, W0, b0, b1, Wl, bl, W1p, tid, wave, lane);
        if (tid < SPW * 4) { int s=tid>>2, c=tid&3;
            av[s][c] += (hs/3.0f)*kk[s][c];
            ys[s][c]  = y0s[s][c] + 0.5f*hs*kk[s][c]; }
        __syncthreads();
        // k3
        rhs_eval(t0 + 0.5f*hs, Xs, ys, kk, red, W0, b0, b1, Wl, bl, W1p, tid, wave, lane);
        if (tid < SPW * 4) { int s=tid>>2, c=tid&3;
            av[s][c] += (hs/3.0f)*kk[s][c];
            ys[s][c]  = y0s[s][c] + hs*kk[s][c]; }
        __syncthreads();
        // k4
        rhs_eval(t0 + hs, Xs, ys, kk, red, W0, b0, b1, Wl, bl, W1p, tid, wave, lane);
        if (tid < SPW * 4) { int s=tid>>2, c=tid&3;
            y0s[s][c] = av[s][c] + (hs/6.0f)*kk[s][c]; }
        __syncthreads();
    }

    if (tid < SPW * 4) {
        int s = tid >> 2, c = tid & 3;
        out[(sbase + s) * 4 + c] = y0s[s][c];
    }
}

// ---------------------------------------------------------------------------
extern "C" void kernel_launch(void* const* d_in, const int* in_sizes, int n_in,
                              void* d_out, int out_size, void* d_ws, size_t ws_size,
                              hipStream_t stream)
{
    const float* states = (const float*)d_in[0];   // (B, 4)
    const float* W0     = (const float*)d_in[1];   // (4, 512)
    const float* b0     = (const float*)d_in[2];   // (512,)
    const float* W1     = (const float*)d_in[3];   // (512, 512)
    const float* b1     = (const float*)d_in[4];   // (512,)
    const float* Wl     = (const float*)d_in[5];   // (512, 3)
    const float* bl     = (const float*)d_in[6];   // (3,)
    float* out = (float*)d_out;

    _Float16* W1p = (_Float16*)d_ws;               // 512 KB packed weights

    // Re-pack every launch (deterministic, graph-capture safe).
    pack_w1<<<(HID * HID) / 256, 256, 0, stream>>>(W1, W1p);

    int B = in_sizes[0] / 4;                       // 65536
    cnf_rk4<<<B / SPW, 256, 0, stream>>>(states, W0, b0, b1, Wl, bl, W1p, out);
}
